// Attention_69947837383104
// MI455X (gfx1250) — matrix operand order repaired and checked by reading, not compiled, not run on hardware
//
#include <hip/hip_runtime.h>

#define D_MODEL 1024
#define NUM_HEADS 16
#define D_HEAD 64
#define SEQ 2048
#define BATCH 2

typedef __attribute__((ext_vector_type(16))) __bf16 v16bf;
typedef __attribute__((ext_vector_type(8)))  float  v8f;

union BV {
    v16bf bf;
    unsigned int u[8];
    uint4 q[2];
};

__device__ __forceinline__ unsigned int f2bf(float f) {
    unsigned int u = __float_as_uint(f);
    return (u + 0x7FFFu + ((u >> 16) & 1u)) >> 16;   // RNE to bf16 bits
}
__device__ __forceinline__ unsigned int pack2(float lo, float hi) {
    return (f2bf(lo) & 0xFFFFu) | (f2bf(hi) << 16);
}
__device__ __forceinline__ v8f vzero() {
    v8f z = {0.f, 0.f, 0.f, 0.f, 0.f, 0.f, 0.f, 0.f};
    return z;
}
__device__ __forceinline__ v8f wmma_bf16(v16bf a, v16bf b, v8f c) {
    return __builtin_amdgcn_wmma_f32_16x16x32_bf16(false, a, false, b, (short)0, c, false, false);
}
__device__ __forceinline__ float fast_exp2(float x) {
    return __builtin_amdgcn_exp2f(x);                // v_exp_f32 (native exp2)
}

// ---------------------------------------------------------------- convert
__global__ __launch_bounds__(256) void cvt_kernel(const float* __restrict__ src,
                                                  unsigned short* __restrict__ dst,
                                                  int n4, float scale) {
    int idx = blockIdx.x * blockDim.x + threadIdx.x;
    if (idx >= n4) return;
    float4 v = reinterpret_cast<const float4*>(src)[idx];
    uint2 o;
    o.x = pack2(v.x * scale, v.y * scale);
    o.y = pack2(v.z * scale, v.w * scale);
    reinterpret_cast<uint2*>(dst)[idx] = o;
}

// ---------------------------------------------------------------- Q/K projection
// C[h][p] = sum_d W[h][d] * X[p][d] ; store to [p][h] (packed 16B per lane)
// wave tile: 64 (h) x 64 (p)  -> 16 WMMA per 8 loads per k-step
__global__ __launch_bounds__(256) void proj_qk_kernel(const unsigned short* __restrict__ xb,
                                                      const unsigned short* __restrict__ wq,
                                                      const unsigned short* __restrict__ wk,
                                                      unsigned short* __restrict__ Qm,
                                                      unsigned short* __restrict__ Km) {
    const int lane = threadIdx.x & 31, wave = threadIdx.x >> 5;
    const int half = lane >> 4, l16 = lane & 15;
    const int yy = blockIdx.y;                 // b*32 + head*2 + which
    const int which = yy & 1, head = (yy >> 1) & (NUM_HEADS - 1), b = yy >> 5;
    const unsigned short* W = which ? wk : wq;
    unsigned short* Out = which ? Km : Qm;
    const int pbase = blockIdx.x * 512 + wave * 64;

    const unsigned short* wrow[4];
#pragma unroll
    for (int mt = 0; mt < 4; ++mt)
        wrow[mt] = W + (size_t)(head * D_HEAD + mt * 16 + l16) * D_MODEL;
    const unsigned short* xrow[4];
#pragma unroll
    for (int nt = 0; nt < 4; ++nt)
        xrow[nt] = xb + (size_t)(b * SEQ + pbase + nt * 16 + l16) * D_MODEL;

    v8f acc[4][4];
#pragma unroll
    for (int mt = 0; mt < 4; ++mt)
#pragma unroll
        for (int nt = 0; nt < 4; ++nt) acc[mt][nt] = vzero();

    for (int kc = 0; kc < D_MODEL; kc += 32) {
        BV a[4], bm[4];
#pragma unroll
        for (int mt = 0; mt < 4; ++mt) {
            a[mt].q[0] = *reinterpret_cast<const uint4*>(wrow[mt] + kc + 16 * half);
            a[mt].q[1] = *reinterpret_cast<const uint4*>(wrow[mt] + kc + 16 * half + 8);
        }
#pragma unroll
        for (int nt = 0; nt < 4; ++nt) {
            const uint4* p = reinterpret_cast<const uint4*>(xrow[nt] + kc + 16 * half);
            bm[nt].q[0] = p[0]; bm[nt].q[1] = p[1];
        }
#pragma unroll
        for (int mt = 0; mt < 4; ++mt)
#pragma unroll
            for (int nt = 0; nt < 4; ++nt)
                acc[mt][nt] = wmma_bf16(a[mt].bf, bm[nt].bf, acc[mt][nt]);
    }
#pragma unroll
    for (int nt = 0; nt < 4; ++nt) {
        const int p = pbase + nt * 16 + l16;
        unsigned short* orow = Out + (size_t)((b * NUM_HEADS + head) * SEQ + p) * D_HEAD;
#pragma unroll
        for (int mt = 0; mt < 4; ++mt) {
            v8f c = acc[mt][nt];
            uint4 o;
            o.x = pack2(c[0], c[1]); o.y = pack2(c[2], c[3]);
            o.z = pack2(c[4], c[5]); o.w = pack2(c[6], c[7]);
            *reinterpret_cast<uint4*>(orow + mt * 16 + 8 * half) = o;
        }
    }
}

// ---------------------------------------------------------------- V projection (stores V^T)
// C[p][h] = sum_d X[p][d] * Wv[h][d] ; store transposed to Vt[h][p]
// wave tile: 64 (p) x 64 (h)
__global__ __launch_bounds__(256) void proj_v_kernel(const unsigned short* __restrict__ xb,
                                                     const unsigned short* __restrict__ wv,
                                                     unsigned short* __restrict__ Vt) {
    const int lane = threadIdx.x & 31, wave = threadIdx.x >> 5;
    const int half = lane >> 4, l16 = lane & 15;
    const int b = blockIdx.y >> 4, head = blockIdx.y & 15;
    const int pbase = blockIdx.x * 512 + wave * 64;

    const unsigned short* xrow[4];
#pragma unroll
    for (int mt = 0; mt < 4; ++mt)
        xrow[mt] = xb + (size_t)(b * SEQ + pbase + mt * 16 + l16) * D_MODEL;
    const unsigned short* wrow[4];
#pragma unroll
    for (int nt = 0; nt < 4; ++nt)
        wrow[nt] = wv + (size_t)(head * D_HEAD + nt * 16 + l16) * D_MODEL;

    v8f acc[4][4];
#pragma unroll
    for (int mt = 0; mt < 4; ++mt)
#pragma unroll
        for (int nt = 0; nt < 4; ++nt) acc[mt][nt] = vzero();

    for (int kc = 0; kc < D_MODEL; kc += 32) {
        BV a[4], bm[4];
#pragma unroll
        for (int mt = 0; mt < 4; ++mt) {
            a[mt].q[0] = *reinterpret_cast<const uint4*>(xrow[mt] + kc + 16 * half);
            a[mt].q[1] = *reinterpret_cast<const uint4*>(xrow[mt] + kc + 16 * half + 8);
        }
#pragma unroll
        for (int nt = 0; nt < 4; ++nt) {
            const uint4* p = reinterpret_cast<const uint4*>(wrow[nt] + kc + 16 * half);
            bm[nt].q[0] = p[0]; bm[nt].q[1] = p[1];
        }
#pragma unroll
        for (int mt = 0; mt < 4; ++mt)
#pragma unroll
            for (int nt = 0; nt < 4; ++nt)
                acc[mt][nt] = wmma_bf16(a[mt].bf, bm[nt].bf, acc[mt][nt]);
    }
#pragma unroll
    for (int nt = 0; nt < 4; ++nt) {
        const int h = nt * 16 + l16;
        unsigned short* orow = Vt + (size_t)((b * NUM_HEADS + head) * D_HEAD + h) * SEQ;
#pragma unroll
        for (int mt = 0; mt < 4; ++mt) {
            v8f c = acc[mt][nt];
            uint4 o;
            o.x = pack2(c[0], c[1]); o.y = pack2(c[2], c[3]);
            o.z = pack2(c[4], c[5]); o.w = pack2(c[6], c[7]);
            *reinterpret_cast<uint4*>(orow + pbase + mt * 16 + 8 * half) = o;
        }
    }
}

// ---------------------------------------------------------------- flash attention (transposed)
// ST[p][q] = sum_h K[p][h]*Q[q][h] (pre-scaled by log2e/sqrt(dh));
// base-2 online softmax per column q; z^T[h][q] += Vt x P^T
// wave tile: 32 q rows (two 16-col groups sharing K/Vt operand loads)
__global__ __launch_bounds__(256) void attn_kernel(const unsigned short* __restrict__ Qm,
                                                   const unsigned short* __restrict__ Km,
                                                   const unsigned short* __restrict__ Vt,
                                                   unsigned short* __restrict__ Zm) {
    const int lane = threadIdx.x & 31, wave = threadIdx.x >> 5;
    const int half = lane >> 4, l16 = lane & 15;
    const int b = blockIdx.y >> 4, head = blockIdx.y & 15;
    const int q0 = blockIdx.x * 256 + wave * 32;
    const size_t bh = (size_t)(b * NUM_HEADS + head);

    // Q is the B operand: per column group qg, lane holds column q = q0 + qg*16 + l16
    BV qb[2][2];
#pragma unroll
    for (int qg = 0; qg < 2; ++qg) {
        const unsigned short* qrow = Qm + (bh * SEQ + q0 + qg * 16 + l16) * D_HEAD;
#pragma unroll
        for (int kc2 = 0; kc2 < 2; ++kc2) {
            const uint4* p = reinterpret_cast<const uint4*>(qrow + kc2 * 32 + 16 * half);
            qb[qg][kc2].q[0] = p[0]; qb[qg][kc2].q[1] = p[1];
        }
    }

    const unsigned short* kbase = Km + bh * SEQ * D_HEAD;
    const unsigned short* vbase = Vt + bh * D_HEAD * SEQ;

    v8f zacc[2][4];
#pragma unroll
    for (int qg = 0; qg < 2; ++qg)
#pragma unroll
        for (int ht = 0; ht < 4; ++ht) zacc[qg][ht] = vzero();
    float m_run[2] = {-3.0e38f, -3.0e38f};
    float l_run[2] = {0.f, 0.f};
    int qv[2];
#pragma unroll
    for (int qg = 0; qg < 2; ++qg) qv[qg] = q0 + qg * 16 + l16;
    const int pend = q0 + 32;                       // causal: only p <= q needed

    for (int pc = 0; pc < pend; pc += 32) {
        // ---- K tiles (shared by both column groups)
        BV ka[2][2];                                // [p-tile][k-chunk]
#pragma unroll
        for (int kc2 = 0; kc2 < 2; ++kc2) {
#pragma unroll
            for (int t = 0; t < 2; ++t) {
                const unsigned short* kr = kbase + (size_t)(pc + t * 16 + l16) * D_HEAD + kc2 * 32;
                ka[t][kc2].q[0] = *reinterpret_cast<const uint4*>(kr + 16 * half);
                ka[t][kc2].q[1] = *reinterpret_cast<const uint4*>(kr + 16 * half + 8);
            }
        }
        // ---- Vt tiles (shared by both column groups)
        BV av[4];
#pragma unroll
        for (int ht = 0; ht < 4; ++ht) {
            const unsigned short* vr = vbase + (size_t)(ht * 16 + l16) * SEQ + pc;
            av[ht].q[0] = *reinterpret_cast<const uint4*>(vr + 16 * half);
            av[ht].q[1] = *reinterpret_cast<const uint4*>(vr + 16 * half + 8);
        }
        // ---- scores^T (already in log2 domain)
        v8f st[2][2];                               // [p-tile][qg]
#pragma unroll
        for (int t = 0; t < 2; ++t)
#pragma unroll
            for (int qg = 0; qg < 2; ++qg) st[t][qg] = vzero();
#pragma unroll
        for (int kc2 = 0; kc2 < 2; ++kc2)
#pragma unroll
            for (int t = 0; t < 2; ++t)
#pragma unroll
                for (int qg = 0; qg < 2; ++qg)
                    st[t][qg] = wmma_bf16(ka[t][kc2].bf, qb[qg][kc2].bf, st[t][qg]);

        const int pb0 = pc + 8 * half, pb1 = pc + 16 + 8 * half;
#pragma unroll
        for (int qg = 0; qg < 2; ++qg) {
            const int q = qv[qg];
            float s0[8], s1[8];
            float mx = -3.0e38f;
#pragma unroll
            for (int r = 0; r < 8; ++r) {
                float v0 = st[0][qg][r]; if (pb0 + r > q) v0 = -1.0e30f;
                float v1 = st[1][qg][r]; if (pb1 + r > q) v1 = -1.0e30f;
                s0[r] = v0; s1[r] = v1;
                mx = fmaxf(mx, fmaxf(v0, v1));
            }
            mx = fmaxf(mx, __shfl_xor(mx, 16));
            const float m_new = fmaxf(m_run[qg], mx);
            const float scale = fast_exp2(m_run[qg] - m_new);
            float lsum = 0.f;
#pragma unroll
            for (int r = 0; r < 8; ++r) {
                s0[r] = fast_exp2(s0[r] - m_new);
                s1[r] = fast_exp2(s1[r] - m_new);
                lsum += s0[r] + s1[r];
            }
            lsum += __shfl_xor(lsum, 16);
            l_run[qg] = l_run[qg] * scale + lsum;
            m_run[qg] = m_new;
#pragma unroll
            for (int ht = 0; ht < 4; ++ht) zacc[qg][ht] = zacc[qg][ht] * scale;

            // build P^T B-operand (K=32 p's, N=16 q's) from score C-layout via xor-16 swaps
            float o0[8], o1[8];
#pragma unroll
            for (int r = 0; r < 8; ++r) {
                o0[r] = __shfl_xor(s0[r], 16);
                o1[r] = __shfl_xor(s1[r], 16);
            }
            BV pb;
#pragma unroll
            for (int jj = 0; jj < 4; ++jj) {        // elements j = 0..7
                float e0 = half ? o1[jj * 2]     : s0[jj * 2];
                float e1 = half ? o1[jj * 2 + 1] : s0[jj * 2 + 1];
                pb.u[jj] = pack2(e0, e1);
            }
#pragma unroll
            for (int jj = 0; jj < 4; ++jj) {        // elements j = 8..15
                float e0 = half ? s1[jj * 2]     : o0[jj * 2];
                float e1 = half ? s1[jj * 2 + 1] : o0[jj * 2 + 1];
                pb.u[4 + jj] = pack2(e0, e1);
            }
            // z^T accumulate: A = Vt rows (h), B = P^T
#pragma unroll
            for (int ht = 0; ht < 4; ++ht)
                zacc[qg][ht] = wmma_bf16(av[ht].bf, pb.bf, zacc[qg][ht]);
        }
    }
#pragma unroll
    for (int qg = 0; qg < 2; ++qg) {
        const float inv = 1.0f / l_run[qg];
#pragma unroll
        for (int ht = 0; ht < 4; ++ht) {
            v8f c = zacc[qg][ht];
            uint4 o;
            o.x = pack2(c[0] * inv, c[1] * inv);
            o.y = pack2(c[2] * inv, c[3] * inv);
            o.z = pack2(c[4] * inv, c[5] * inv);
            o.w = pack2(c[6] * inv, c[7] * inv);
            unsigned short* zr = Zm + (size_t)(b * SEQ + qv[qg]) * D_MODEL
                               + head * D_HEAD + ht * 16 + 8 * half;
            *reinterpret_cast<uint4*>(zr) = o;
        }
    }
}

// ---------------------------------------------------------------- output projection (fp32 out)
// out[m][d] = sum_f Z[m][f] * W_O[d][f] ; wave tile 64 x 64
__global__ __launch_bounds__(256) void oproj_kernel(const unsigned short* __restrict__ Zm,
                                                    const unsigned short* __restrict__ wo,
                                                    float* __restrict__ out) {
    const int lane = threadIdx.x & 31, wave = threadIdx.x >> 5;
    const int half = lane >> 4, l16 = lane & 15;
    const int mbase = blockIdx.y * 128 + (wave & 1) * 64;
    const int nbase = blockIdx.x * 256 + (wave >> 1) * 64;

    const unsigned short* zrow[4];
#pragma unroll
    for (int mt = 0; mt < 4; ++mt)
        zrow[mt] = Zm + (size_t)(mbase + mt * 16 + l16) * D_MODEL;
    const unsigned short* wrow[4];
#pragma unroll
    for (int nt = 0; nt < 4; ++nt)
        wrow[nt] = wo + (size_t)(nbase + nt * 16 + l16) * D_MODEL;

    v8f acc[4][4];
#pragma unroll
    for (int mt = 0; mt < 4; ++mt)
#pragma unroll
        for (int nt = 0; nt < 4; ++nt) acc[mt][nt] = vzero();

    for (int kc = 0; kc < D_MODEL; kc += 32) {
        BV a[4], bm[4];
#pragma unroll
        for (int mt = 0; mt < 4; ++mt) {
            a[mt].q[0] = *reinterpret_cast<const uint4*>(zrow[mt] + kc + 16 * half);
            a[mt].q[1] = *reinterpret_cast<const uint4*>(zrow[mt] + kc + 16 * half + 8);
        }
#pragma unroll
        for (int nt = 0; nt < 4; ++nt) {
            const uint4* p = reinterpret_cast<const uint4*>(wrow[nt] + kc + 16 * half);
            bm[nt].q[0] = p[0]; bm[nt].q[1] = p[1];
        }
#pragma unroll
        for (int mt = 0; mt < 4; ++mt)
#pragma unroll
            for (int nt = 0; nt < 4; ++nt)
                acc[mt][nt] = wmma_bf16(a[mt].bf, bm[nt].bf, acc[mt][nt]);
    }
#pragma unroll
    for (int mt = 0; mt < 4; ++mt)
#pragma unroll
        for (int nt = 0; nt < 4; ++nt) {
            const int d = nbase + nt * 16 + l16;
#pragma unroll
            for (int r = 0; r < 8; ++r) {
                const int m = mbase + mt * 16 + 8 * half + r;
                out[(size_t)m * D_MODEL + d] = acc[mt][nt][r];
            }
        }
}

// ---------------------------------------------------------------- host launcher
extern "C" void kernel_launch(void* const* d_in, const int* in_sizes, int n_in,
                              void* d_out, int out_size, void* d_ws, size_t ws_size,
                              hipStream_t stream) {
    const float* x   = (const float*)d_in[0];
    const float* W_K = (const float*)d_in[1];
    const float* W_Q = (const float*)d_in[2];
    const float* W_V = (const float*)d_in[3];
    const float* W_O = (const float*)d_in[4];
    float* out = (float*)d_out;

    // workspace layout (bf16 halves), total = 48 MB
    const size_t NX = (size_t)BATCH * SEQ * D_MODEL;          // 4 Mi elems
    const size_t NW = (size_t)NUM_HEADS * D_HEAD * D_MODEL;   // 1 Mi elems
    unsigned short* ws = (unsigned short*)d_ws;
    unsigned short* xb = ws; ws += NX;
    unsigned short* wq = ws; ws += NW;
    unsigned short* wk = ws; ws += NW;
    unsigned short* wv = ws; ws += NW;
    unsigned short* wo = ws; ws += NW;
    unsigned short* Qm = ws; ws += NX;
    unsigned short* Km = ws; ws += NX;
    unsigned short* Vt = ws; ws += NX;
    unsigned short* Zm = ws; ws += NX;

    const int T = 256;
    // Q scale = (1/sqrt(d_head)) * log2(e)  -> base-2 softmax domain
    const float qscale = 0.125f * 1.4426950408889634f;
    cvt_kernel<<<(int)((NX / 4 + T - 1) / T), T, 0, stream>>>(x,   xb, (int)(NX / 4), 1.0f);
    cvt_kernel<<<(int)((NW / 4 + T - 1) / T), T, 0, stream>>>(W_Q, wq, (int)(NW / 4), qscale);
    cvt_kernel<<<(int)((NW / 4 + T - 1) / T), T, 0, stream>>>(W_K, wk, (int)(NW / 4), 1.0f);
    cvt_kernel<<<(int)((NW / 4 + T - 1) / T), T, 0, stream>>>(W_V, wv, (int)(NW / 4), 1.0f);
    cvt_kernel<<<(int)((NW / 4 + T - 1) / T), T, 0, stream>>>(W_O, wo, (int)(NW / 4), 1.0f);

    proj_qk_kernel<<<dim3(4, 64), 256, 0, stream>>>(xb, wq, wk, Qm, Km);
    proj_v_kernel <<<dim3(4, 32), 256, 0, stream>>>(xb, wv, Vt);
    attn_kernel   <<<dim3(8, 32), 256, 0, stream>>>(Qm, Km, Vt, Zm);
    oproj_kernel  <<<dim3(4, 32), 256, 0, stream>>>(Zm, wo, out);
}
